// A2C2f_5446018531988
// MI455X (gfx1250) — matrix operand order repaired and checked by reading, not compiled, not run on hardware
//
#include <hip/hip_runtime.h>
#include <hip/hip_bf16.h>

typedef __attribute__((ext_vector_type(16))) __bf16 v16bf;
typedef __attribute__((ext_vector_type(8)))  __bf16 v8bf;
typedef __attribute__((ext_vector_type(8)))  float  v8f;

#define HWPIX 2304   // 48*48 pixels per image
#define NB    4      // batch

// ---------- bf16 helpers ----------
static __device__ __forceinline__ __bf16 f2bf(float f) {
  unsigned u = __builtin_bit_cast(unsigned, f);
  unsigned r = (u + 0x7FFFu + ((u >> 16) & 1u)) >> 16;   // RTNE
  unsigned short s = (unsigned short)r;
  return __builtin_bit_cast(__bf16, s);
}
static __device__ __forceinline__ float bf2f(__bf16 b) {
  unsigned short s = __builtin_bit_cast(unsigned short, b);
  unsigned u = ((unsigned)s) << 16;
  return __builtin_bit_cast(float, u);
}
static __device__ __forceinline__ unsigned pk_bf16(float a, float b) {
  unsigned ua = __builtin_bit_cast(unsigned, a);
  unsigned ub = __builtin_bit_cast(unsigned, b);
  ua = (ua + 0x7FFFu + ((ua >> 16) & 1u)) >> 16;
  ub = (ub + 0x7FFFu + ((ub >> 16) & 1u)) & 0xFFFF0000u;
  return ua | ub;
}
static __device__ __forceinline__ float fast_silu(float v) {
  float e = __expf(-v);
  return v * __builtin_amdgcn_rcpf(1.0f + e);
}
static __device__ __forceinline__ v8f wmma_bf16(v16bf a, v16bf b, v8f c) {
  return __builtin_amdgcn_wmma_f32_16x16x32_bf16(false, a, false, b, (short)0, c, false, false);
}
// A-fragment: element i -> K = (i&7) + 16*(i>=8) + 8*(lane>=16); p at K-offset 8*(lane>=16)
static __device__ __forceinline__ v16bf load_a_frag(const __bf16* p) {
  v16bf r;
#pragma unroll
  for (int i = 0; i < 8; ++i) { r[i] = p[i]; r[i + 8] = p[i + 16]; }
  return r;
}
// B-fragment: element i -> K = (i&7) + 16*(i>=8) + 8*(lane>=16); p at K-offset 8*(lane>=16)
static __device__ __forceinline__ v16bf load_b_frag(const __bf16* p) {
  v16bf r;
#pragma unroll
  for (int i = 0; i < 8; ++i) { r[i] = p[i]; r[i + 8] = p[i + 16]; }
  return r;
}
// flat-pointer low 32 bits are the LDS byte offset (ISA 10.2 aperture mapping)
static __device__ __forceinline__ unsigned lds_off32(const void* p) {
  return (unsigned)(unsigned long long)p;
}
// CDNA5 async global->LDS copy, 16B per lane, tracked by ASYNCcnt
static __device__ __forceinline__ void async_copy_b128(unsigned lds_byte_off, const void* gptr) {
  asm volatile("global_load_async_to_lds_b128 %0, %1, off"
               :: "v"(lds_byte_off), "v"(gptr) : "memory");
}
template <int N>
static __device__ __forceinline__ void wait_async() {
  asm volatile("s_wait_asynccnt %0" :: "i"(N) : "memory");
}

// ---------- f32 -> bf16 bulk conversion (weights, row-major preserved) ----------
__global__ __launch_bounds__(256) void cvt_f32_bf16_k(const float* __restrict__ src,
                                                      __bf16* __restrict__ dst, int n4) {
  int i = blockIdx.x * 256 + threadIdx.x;
  if (i >= n4) return;
  float4 f = ((const float4*)src)[i];
  uint2 o;
  o.x = pk_bf16(f.x, f.y);
  o.y = pk_bf16(f.z, f.w);
  ((uint2*)dst)[i] = o;
}

// ---------- x: NCHW f32 -> pixel-major [b][p][c] bf16 ----------
__global__ __launch_bounds__(256) void cvt_x_k(const float* __restrict__ x,
                                               __bf16* __restrict__ xbf) {
  int t = blockIdx.x * 256 + threadIdx.x;        // 0..NB*HWPIX-1
  int b = t / HWPIX, p = t - b * HWPIX;
  const float* s = x + (size_t)b * 512 * HWPIX + p;
  unsigned* d = (unsigned*)(xbf + (size_t)t * 512);
#pragma unroll 4
  for (int c = 0; c < 512; c += 2)
    d[c >> 1] = pk_bf16(s[(size_t)c * HWPIX], s[(size_t)(c + 1) * HWPIX]);
}

// ---------- bf16 WMMA GEMM: W row-major [O][C], X pixel-major [p][Xc] ----------
// Double-buffered async global->LDS staging: next K-step prefetched under WMMA,
// fenced with the partial-count s_wait_asynccnt (ASYNC ops complete in order).
__global__ __launch_bounds__(256) void gemm_bf16_k(
    const __bf16* __restrict__ Wbf, const float* __restrict__ bias,
    const __bf16* __restrict__ X, int Xc, const __bf16* __restrict__ X2, int X2c, int csplit,
    float* __restrict__ Yf, __bf16* __restrict__ Ybf,
    const float* __restrict__ res, const float* __restrict__ scale,
    int O, int C, int act)
{
  __shared__ __align__(16) __bf16 ldsA[2][64][32];    // W tiles, ping-pong (8KB)
  __shared__ __align__(16) __bf16 ldsB[2][128][32];   // X tiles, ping-pong (16KB)

  const int tid = threadIdx.x;
  const int z   = blockIdx.z;
  const int m0  = blockIdx.y * 64;
  const int n0  = blockIdx.x * 128;

  const __bf16* Xz  = X + (size_t)z * HWPIX * Xc;
  const __bf16* X2z = X2 ? X2 + (size_t)z * HWPIX * X2c : nullptr;
  float*  Yfz  = Yf  ? Yf  + (size_t)z * O * HWPIX : nullptr;
  __bf16* Ybfz = Ybf ? Ybf + (size_t)z * HWPIX * O : nullptr;
  const float* Rz = res ? res + (size_t)z * O * HWPIX : nullptr;

  const int lane = tid & 31;
  const int wv   = tid >> 5;
  const int mw   = wv >> 2;          // 0..1
  const int nw   = wv & 3;           // 0..3
  const int lhi  = lane >> 4, lm = lane & 15;

  const unsigned ldsA_off = lds_off32(&ldsA[0][0][0]);
  const unsigned ldsB_off = lds_off32(&ldsB[0][0][0]);

  const int am = tid >> 2, aq = tid & 3;           // A: 64 rows x 4 chunks
  const int bn = tid >> 1, bq0 = (tid & 1) * 2;    // B: 128 rows x 4 chunks, 2/thread

  // per-thread staging: 1 A-chunk + 2 B-chunks per K-step (3 async ops)
  auto stage = [&](int buf, int kc) {
    async_copy_b128(ldsA_off + (unsigned)(buf * 4096 + am * 64 + aq * 16),
                    Wbf + (size_t)(m0 + am) * C + kc + aq * 8);
#pragma unroll
    for (int c = 0; c < 2; ++c) {
      int q  = bq0 + c;
      int kk = kc + q * 8;
      const __bf16* src = (kk < csplit)
          ? (Xz  + (size_t)(n0 + bn) * Xc  + kk)
          : (X2z + (size_t)(n0 + bn) * X2c + (kk - csplit));
      async_copy_b128(ldsB_off + (unsigned)(buf * 8192 + bn * 64 + q * 16), src);
    }
  };

  v8f acc[2][2];
#pragma unroll
  for (int a = 0; a < 2; ++a)
#pragma unroll
    for (int b = 0; b < 2; ++b)
#pragma unroll
      for (int i = 0; i < 8; ++i) acc[a][b][i] = 0.f;

  stage(0, 0);
  int cur = 0;
  for (int kc = 0; kc < C; kc += 32) {
    if (kc + 32 < C) {
      stage(cur ^ 1, kc + 32);   // prefetch next K-step into the other buffer
      wait_async<3>();           // wait only for the OLDER 3 ops (current buffer)
    } else {
      wait_async<0>();
    }
    __syncthreads();

    v16bf afrag[2], bfrag[2];
#pragma unroll
    for (int ms = 0; ms < 2; ++ms)
      afrag[ms] = load_a_frag(&ldsA[cur][mw * 32 + ms * 16 + lm][8 * lhi]);
#pragma unroll
    for (int ns = 0; ns < 2; ++ns)
      bfrag[ns] = load_b_frag(&ldsB[cur][nw * 32 + ns * 16 + lm][8 * lhi]);
#pragma unroll
    for (int ms = 0; ms < 2; ++ms)
#pragma unroll
      for (int ns = 0; ns < 2; ++ns)
        acc[ms][ns] = wmma_bf16(afrag[ms], bfrag[ns], acc[ms][ns]);
    __syncthreads();             // orders next prefetch writes vs this step's reads
    cur ^= 1;
  }

  // epilogue: element r -> M = r + 8*(lane>=16), N = lane&15
#pragma unroll
  for (int ms = 0; ms < 2; ++ms) {
#pragma unroll
    for (int ns = 0; ns < 2; ++ns) {
      const int obase = m0 + mw * 32 + ms * 16 + 8 * lhi;     // multiple of 8
      const int p     = n0 + nw * 32 + ns * 16 + lm;
      float t[8];
#pragma unroll
      for (int r = 0; r < 8; ++r) {
        int oo = obase + r;
        float v = acc[ms][ns][r] + bias[oo];
        if (act)   v = fast_silu(v);
        if (scale) v *= scale[oo];
        if (Rz)    v += Rz[(size_t)oo * HWPIX + p];
        if (Yfz)   Yfz[(size_t)oo * HWPIX + p] = v;
        t[r] = v;
      }
      if (Ybfz) {
        unsigned* bp = (unsigned*)(Ybfz + (size_t)p * O + obase);
#pragma unroll
        for (int rp = 0; rp < 4; ++rp) bp[rp] = pk_bf16(t[2 * rp], t[2 * rp + 1]);
      }
    }
  }
}

// ---------- area attention (pixel-major bf16 qkv in, pixel-major bf16 out) ----------
// one WG per (area-batch, head, 64-query block); K/V/S resident in 144KB LDS
__global__ __launch_bounds__(256) void area_attn_k(
    const __bf16* __restrict__ qkv, __bf16* __restrict__ attn_out)
{
  extern __shared__ __align__(16) char smem_raw[];
  __bf16* ldsK = (__bf16*)smem_raw;          // [576][32]  K (m-major) - direct async
  __bf16* ldsV = ldsK + 576 * 32;            // [32][576]  V (d-major)
  __bf16* ldsS = ldsV + 32 * 576;            // [64][576]  V-scratch -> scores -> probs

  const int tid  = threadIdx.x;
  const int lane = tid & 31, wv = tid >> 5;
  const int lhi  = lane >> 4, lm = lane & 15;

  const int ba   = blockIdx.z;               // 0..15
  const int b    = ba >> 2, area = ba & 3;
  const int h    = blockIdx.y;               // 0..7
  const int nb0  = blockIdx.x * 64;

  const __bf16* qkvb = qkv + (size_t)b * HWPIX * 768;   // pixel-major rows of 768 ch
  const int pbase = area * 576;

  const unsigned ldsK_off = lds_off32(ldsK);
  const unsigned ldsS_off = lds_off32(ldsS);

  // phase 1a: async K -> ldsK [m][d] (final layout!), V -> ldsS scratch [m][d]
#pragma unroll
  for (int it = 0; it < 9; ++it) {
    int cc = tid + it * 256;                 // 2304 chunks of 16B
    int m  = cc >> 2;
    int q  = cc & 3;
    const __bf16* row = qkvb + (size_t)(pbase + m) * 768 + h * 96;
    async_copy_b128(ldsK_off + (unsigned)(m * 64 + q * 16), row + 32 + q * 8);
    async_copy_b128(ldsS_off + (unsigned)(m * 64 + q * 16), row + 64 + q * 8);
  }
  wait_async<0>();
  __syncthreads();

  // phase 1b: transpose V scratch [m][d] -> ldsV [d][m]
#pragma unroll
  for (int it = 0; it < 9; ++it) {
    int cc = tid + it * 256;
    int m  = cc >> 2;
    int d0 = (cc & 3) * 8;
    v8bf vv = *(const v8bf*)(ldsS + m * 32 + d0);
#pragma unroll
    for (int j = 0; j < 8; ++j) ldsV[(d0 + j) * 576 + m] = vv[j];
  }
  __syncthreads();

  // phase 2: S[n][m] = (q^T k)[n][m] * hd^-0.5  (one WMMA per tile, K=hd=32)
  {
    const int ntile = wv & 3, mhalf = wv >> 2;
    const int n = nb0 + ntile * 16 + lm;
    const __bf16* qp = qkvb + (size_t)(pbase + n) * 768 + h * 96 + 8 * lhi;
    v8bf q0 = *(const v8bf*)qp;          // d = 8*lhi .. +7
    v8bf q1 = *(const v8bf*)(qp + 16);   // d = 16+8*lhi .. +7
    v16bf qa;
#pragma unroll
    for (int j = 0; j < 8; ++j) { qa[j] = q0[j]; qa[j + 8] = q1[j]; }
    const float rsc = 0.17677669529663689f;               // 32^-0.5
    v8f zero8;
#pragma unroll
    for (int i = 0; i < 8; ++i) zero8[i] = 0.f;
    for (int mt = mhalf * 18; mt < mhalf * 18 + 18; ++mt) {
      v16bf kb = load_b_frag(&ldsK[(mt * 16 + lm) * 32 + 8 * lhi]);
      v8f s = wmma_bf16(qa, kb, zero8);
      const int m = mt * 16 + lm;
#pragma unroll
      for (int r = 0; r < 8; ++r) {
        int nl = ntile * 16 + r + 8 * lhi;
        ldsS[nl * 576 + m] = f2bf(s[r] * rsc);
      }
    }
  }
  __syncthreads();

  // phase 3: row softmax over m (576), wave32 shuffle reductions, in-place bf16
  for (int rr = 0; rr < 8; ++rr) {
    const int row = wv * 8 + rr;
    __bf16* srow = ldsS + row * 576;
    float mx = -3.4e38f;
    for (int m = lane; m < 576; m += 32) mx = fmaxf(mx, bf2f(srow[m]));
#pragma unroll
    for (int off = 16; off > 0; off >>= 1) mx = fmaxf(mx, __shfl_xor(mx, off, 32));
    float sum = 0.f;
    for (int m = lane; m < 576; m += 32) {
      float e = __expf(bf2f(srow[m]) - mx);
      srow[m] = f2bf(e);
      sum += e;
    }
#pragma unroll
    for (int off = 16; off > 0; off >>= 1) sum += __shfl_xor(sum, off, 32);
    const float inv = __builtin_amdgcn_rcpf(sum);
    for (int m = lane; m < 576; m += 32) srow[m] = f2bf(bf2f(srow[m]) * inv);
  }
  __syncthreads();

  // phase 4: out[n][d] = sum_m P[n][m] * v[d][m]  (18 WMMA K-steps)
  {
    const int ntile = wv & 3, dtile = wv >> 2;
    v8f acc;
#pragma unroll
    for (int i = 0; i < 8; ++i) acc[i] = 0.f;
    const int nrow = ntile * 16 + lm;
    for (int kc = 0; kc < 576; kc += 32) {
      v16bf pa = load_a_frag(&ldsS[nrow * 576 + kc + 8 * lhi]);
      v16bf vb = load_b_frag(&ldsV[(dtile * 16 + lm) * 576 + kc + 8 * lhi]);
      acc = wmma_bf16(pa, vb, acc);
    }
    // pixel-major store: element r -> pixel pbase+nb0+ntile*16+8*lhi+r, channel c
    const int c    = h * 32 + dtile * 16 + lm;
    const int pix0 = pbase + nb0 + ntile * 16 + 8 * lhi;
    __bf16* op = attn_out + ((size_t)b * HWPIX + pix0) * 256 + c;
#pragma unroll
    for (int r = 0; r < 8; ++r) op[(size_t)r * 256] = f2bf(acc[r]);
  }
}

// ---------- depthwise 7x7 positional conv on v (bf16), accumulate into attn (bf16) ----------
__global__ __launch_bounds__(256) void dwpe_k(
    const __bf16* __restrict__ qkv, const float* __restrict__ wpe,
    const float* __restrict__ bpe, __bf16* __restrict__ attn)
{
  const int pix = blockIdx.x * 256 + threadIdx.x;  // 0..2303
  const int c   = blockIdx.y;
  const int b   = blockIdx.z;
  const int hh = pix / 48, ww = pix - hh * 48;
  const int qc = (c >> 5) * 96 + 64 + (c & 31);    // v-channel inside qkv
  const __bf16* src = qkv + (size_t)b * HWPIX * 768 + qc;   // pixel-major: [p*768 + qc]
  const float* wp = wpe + c * 49;
  float acc = bpe[c];
#pragma unroll
  for (int ky = 0; ky < 7; ++ky) {
    int y = hh + ky - 3;
    if (y < 0 || y >= 48) continue;
#pragma unroll
    for (int kx = 0; kx < 7; ++kx) {
      int x = ww + kx - 3;
      if (x < 0 || x >= 48) continue;
      acc += bf2f(src[(size_t)(y * 48 + x) * 768]) * wp[ky * 7 + kx];
    }
  }
  __bf16* dst = attn + ((size_t)b * HWPIX + pix) * 256 + c;
  *dst = f2bf(bf2f(*dst) + acc);
}

// ---------- host orchestration ----------
extern "C" void kernel_launch(void* const* d_in, const int* in_sizes, int n_in,
                              void* d_out, int out_size, void* d_ws, size_t ws_size,
                              hipStream_t stream) {
  const float* x      = (const float*)d_in[0];
  const float* w_cv1  = (const float*)d_in[1];
  const float* b_cv1  = (const float*)d_in[2];
  const float* w_qkv  = (const float*)d_in[3];
  const float* b_qkv  = (const float*)d_in[4];
  const float* w_proj = (const float*)d_in[5];
  const float* b_proj = (const float*)d_in[6];
  const float* w_pe   = (const float*)d_in[7];
  const float* b_pe   = (const float*)d_in[8];
  const float* w_m1   = (const float*)d_in[9];
  const float* b_m1   = (const float*)d_in[10];
  const float* w_m2   = (const float*)d_in[11];
  const float* b_m2   = (const float*)d_in[12];
  const float* w_cv2  = (const float*)d_in[13];
  const float* b_cv2  = (const float*)d_in[14];
  const float* gamma  = (const float*)d_in[15];
  float* out = (float*)d_out;

  const size_t P256 = (size_t)NB * 256 * HWPIX;   // 2,359,296
  const size_t P512 = (size_t)NB * 512 * HWPIX;   // 4,718,592
  const size_t P768 = (size_t)NB * 768 * HWPIX;   // 7,077,888

  float*  yf    = (float*)d_ws;                   // f32 running y, NCHW
  __bf16* xbf   = (__bf16*)(yf + P256);           // pixel-major [b][p][512]
  __bf16* y0bf  = xbf + P512;                     // [b][p][256]
  __bf16* ybf   = y0bf + P256;                    // [b][p][256]
  __bf16* qkvbf = ybf + P256;                     // [b][p][768]
  __bf16* hbf   = qkvbf + P768;                   // [b][p][512]
  __bf16* attnbf= hbf + P512;                     // [b][p][256]
  __bf16* wbf   = attnbf + P256;                  // weight arena (row-major bf16)

  __bf16* wb_cv1  = wbf;                 // 131072
  __bf16* wb_qkv  = wb_cv1 + 131072;     // 393216
  __bf16* wb_proj = wb_qkv + 393216;     // 131072
  __bf16* wb_m1   = wb_proj + 131072;    // 262144
  __bf16* wb_m2   = wb_m1 + 262144;      // 262144
  __bf16* wb_cv2  = wb_m2 + 262144;      // 262144

  dim3 blk(256);

  // one-time conversions to bf16
  cvt_f32_bf16_k<<<dim3(131072 / 4 / 256), blk, 0, stream>>>(w_cv1, wb_cv1, 131072 / 4);
  cvt_f32_bf16_k<<<dim3(393216 / 4 / 256), blk, 0, stream>>>(w_qkv, wb_qkv, 393216 / 4);
  cvt_f32_bf16_k<<<dim3(131072 / 4 / 256), blk, 0, stream>>>(w_proj, wb_proj, 131072 / 4);
  cvt_f32_bf16_k<<<dim3(262144 / 4 / 256), blk, 0, stream>>>(w_m1, wb_m1, 262144 / 4);
  cvt_f32_bf16_k<<<dim3(262144 / 4 / 256), blk, 0, stream>>>(w_m2, wb_m2, 262144 / 4);
  cvt_f32_bf16_k<<<dim3(262144 / 4 / 256), blk, 0, stream>>>(w_cv2, wb_cv2, 262144 / 4);
  cvt_x_k<<<dim3(NB * HWPIX / 256), blk, 0, stream>>>(x, xbf);   // NCHW -> pixel-major

  // y0 = silu(cv1(x)); yf = y0 (f32 NCHW), y0bf/ybf pixel-major mirrors
  gemm_bf16_k<<<dim3(18, 4, NB), blk, 0, stream>>>(
      wb_cv1, b_cv1, xbf, 512, nullptr, 0, 512, yf, y0bf, nullptr, nullptr, 256, 512, 1);
  hipMemcpyAsync(ybf, y0bf, P256 * sizeof(__bf16), hipMemcpyDeviceToDevice, stream);

  for (int i = 0; i < 2; ++i) {
    // qkv = Wqkv*y + b  (bf16 pixel-major only)
    gemm_bf16_k<<<dim3(18, 12, NB), blk, 0, stream>>>(
        wb_qkv + (size_t)i * 768 * 256, b_qkv + i * 768,
        ybf, 256, nullptr, 0, 256, nullptr, qkvbf, nullptr, nullptr, 768, 256, 0);
    // area attention
    area_attn_k<<<dim3(9, 8, 16), blk, 147456, stream>>>(qkvbf, attnbf);
    // attn += dwconv7(v) + bpe
    dwpe_k<<<dim3(9, 256, NB), blk, 0, stream>>>(
        qkvbf, w_pe + (size_t)i * 256 * 49, b_pe + i * 256, attnbf);
    // y = y + proj(attn)   (f32 chain + bf16 mirror)
    gemm_bf16_k<<<dim3(18, 4, NB), blk, 0, stream>>>(
        wb_proj + (size_t)i * 256 * 256, b_proj + i * 256,
        attnbf, 256, nullptr, 0, 256, yf, ybf, yf, nullptr, 256, 256, 0);
    // h = silu(mlp1(y));  y = y + mlp2(h)
    gemm_bf16_k<<<dim3(18, 8, NB), blk, 0, stream>>>(
        wb_m1 + (size_t)i * 512 * 256, b_m1 + i * 512,
        ybf, 256, nullptr, 0, 256, nullptr, hbf, nullptr, nullptr, 512, 256, 1);
    gemm_bf16_k<<<dim3(18, 4, NB), blk, 0, stream>>>(
        wb_m2 + (size_t)i * 256 * 512, b_m2 + i * 256,
        hbf, 512, nullptr, 0, 512, yf, ybf, yf, nullptr, 256, 512, 0);
  }

  // out = x + gamma * silu(cv2(concat([y0, y])))
  gemm_bf16_k<<<dim3(18, 8, NB), blk, 0, stream>>>(
      wb_cv2, b_cv2, y0bf, 256, ybf, 256, 256, out, nullptr, x, gamma, 512, 512, 1);
}
